// SparseConvNeXtBlock3d_16527034155711
// MI455X (gfx1250) — matrix operand order repaired and checked by reading, not compiled, not run on hardware
//
#include <hip/hip_runtime.h>

typedef _Float16 v16h __attribute__((ext_vector_type(16)));
typedef float    v8f  __attribute__((ext_vector_type(8)));
typedef int      i32x4 __attribute__((vector_size(16)));

#define NV  262144
#define CC  128
#define HD  512

union FragU { v16h v; float4 f4[2]; };
union Pack16 { _Float16 h[16]; float4 f4[2]; };

// B-fragment: lane = N col, halves = K rows [0..7] and [16..23] (stride ld) relative to (kstep, lane-half) base
static __device__ __forceinline__ v16h load16(const _Float16* p, int ld) {
  v16h v;
#pragma unroll
  for (int s = 0; s < 16; ++s) v[s] = p[(s + (s & 8)) * ld];
  return v;
}

// A-fragment: halves [0..7] and [16..23] relative to (row, kstep, lane-half) base
static __device__ __forceinline__ v16h loadA(const _Float16* p) {
  FragU u;
  u.f4[0] = *(const float4*)p;
  u.f4[1] = *(const float4*)(p + 16);
  return u.v;
}

static __device__ __forceinline__ v8f wmma16(v16h a, v16h b, v8f c) {
  return __builtin_amdgcn_wmma_f32_16x16x32_f16(false, a, false, b, (short)0, c,
                                                false, false);
}

// ---- CDNA5 async memory->LDS copy (ASYNCcnt) -------------------------------
static __device__ __forceinline__ void async_copy_b128(const _Float16* g,
                                                       _Float16* l) {
#if defined(__has_builtin) && __has_builtin(__builtin_amdgcn_global_load_async_to_lds_b128)
  __builtin_amdgcn_global_load_async_to_lds_b128((i32x4*)g, (i32x4*)l, 0, 0);
#else
  unsigned loff = (unsigned)(unsigned long long)(char*)l;
  asm volatile("global_load_async_to_lds_b128 %0, %1, off"
               :: "v"(loff), "v"(g) : "memory");
#endif
}

static __device__ __forceinline__ void wait_async0() {
#if defined(__has_builtin) && __has_builtin(__builtin_amdgcn_s_wait_asynccnt)
  __builtin_amdgcn_s_wait_asynccnt(0);
#else
  asm volatile("s_wait_asynccnt 0x0" ::: "memory");
#endif
}

// ---------------- prep: weight transpose + f32->f16 ----------------
__global__ void k_prep(const float* __restrict__ conv_w,
                       const float* __restrict__ w1,
                       const float* __restrict__ w2,
                       _Float16* __restrict__ wk,
                       _Float16* __restrict__ w1h,
                       _Float16* __restrict__ w2h) {
  int i = blockIdx.x * 256 + threadIdx.x;
  const int NWK = 27 * CC * CC;           // 442368
  if (i < NWK) {
    int k = i >> 14, r = i & 16383;
    int ci = r >> 7, co = r & 127;
    wk[i] = (_Float16)conv_w[co * (27 * CC) + k * CC + ci];
  } else if (i < NWK + CC * HD) {
    int j = i - NWK;
    w1h[j] = (_Float16)w1[j];
  } else if (i < NWK + 2 * CC * HD) {
    int j = i - NWK - CC * HD;
    w2h[j] = (_Float16)w2[j];
  }
}

// ---------------- prep: feats f32 -> f16 (bandwidth bound) ----------------
__global__ __launch_bounds__(256) void k_feats16(const float* __restrict__ f,
                                                 _Float16* __restrict__ o) {
  long i = ((long)blockIdx.x * 256 + threadIdx.x) * 16;
  const float4* s = (const float4*)(f + i);
  Pack16 p;
#pragma unroll
  for (int q = 0; q < 4; ++q) {
    float4 v = s[q];
    p.h[q * 4 + 0] = (_Float16)v.x; p.h[q * 4 + 1] = (_Float16)v.y;
    p.h[q * 4 + 2] = (_Float16)v.z; p.h[q * 4 + 3] = (_Float16)v.w;
  }
  float4* d = (float4*)(o + i);
  d[0] = p.f4[0];
  d[1] = p.f4[1];
}

// ---------------- conv + bias + layernorm, async double-buffered gather ----
__global__ __launch_bounds__(256) void k_conv_ln_async(
    const _Float16* __restrict__ feats16, const int* __restrict__ nbr,
    const _Float16* __restrict__ wk, const float* __restrict__ conv_b,
    const float* __restrict__ ln_g, const float* __restrict__ ln_b,
    _Float16* __restrict__ h16) {
  __shared__ __align__(16) _Float16 Atile[2][32 * CC];  // 2 x 8 KB
  __shared__ __align__(16) float    Htile[32 * CC];     // 16 KB

  const int t = threadIdx.x;
  const int lane = t & 31;
  const int w = t >> 5;                 // wave id == N-tile (0..7)
  const int vox0 = blockIdx.x * 32;

  v8f c0 = {}; v8f c1 = {};

  const int grow = t >> 3;              // gather row 0..31
  const int gseg = t & 7;               // 16 halves (32 B) each

  // issue gather for offset 0 into buffer 0
  {
    int gi = nbr[vox0 + grow];
    _Float16* dst = &Atile[0][grow * CC + gseg * 16];
    if (gi >= 0) {
      const _Float16* src = feats16 + (long)gi * CC + gseg * 16;
      async_copy_b128(src, dst);
      async_copy_b128(src + 8, dst + 8);
    } else {
      float4 z = make_float4(0.f, 0.f, 0.f, 0.f);
      ((float4*)dst)[0] = z;
      ((float4*)dst)[1] = z;
    }
  }

  const int arow = lane & 15;
  const int kh = (lane >> 4) * 8;

  for (int off = 0; off < 27; ++off) {
    const int cur = off & 1;
    wait_async0();
    __syncthreads();                    // gather(off) visible; buf[cur^1] free

    if (off < 26) {                     // prefetch next offset into other buf
      int gi = nbr[(off + 1) * NV + vox0 + grow];
      _Float16* dst = &Atile[cur ^ 1][grow * CC + gseg * 16];
      if (gi >= 0) {
        const _Float16* src = feats16 + (long)gi * CC + gseg * 16;
        async_copy_b128(src, dst);
        async_copy_b128(src + 8, dst + 8);
      } else {
        float4 z = make_float4(0.f, 0.f, 0.f, 0.f);
        ((float4*)dst)[0] = z;
        ((float4*)dst)[1] = z;
      }
    }

    const _Float16* At = Atile[cur];
    const _Float16* wko = wk + off * (CC * CC);
#pragma unroll
    for (int ks = 0; ks < 4; ++ks) {
      v16h b  = load16(wko + (ks * 32 + kh) * CC + w * 16 + (lane & 15), CC);
      v16h a0 = loadA(&At[arow * CC + ks * 32 + kh]);
      v16h a1 = loadA(&At[(arow + 16) * CC + ks * 32 + kh]);
      c0 = wmma16(a0, b, c0);
      c1 = wmma16(a1, b, c1);
    }
  }

  // bias + dump conv result to LDS (f32) for layernorm
  {
    int ncol = w * 16 + (lane & 15);
    int mb = (lane >> 4) * 8;
    float bias = conv_b[ncol];
#pragma unroll
    for (int g = 0; g < 8; ++g) {
      Htile[(mb + g) * CC + ncol]      = c0[g] + bias;
      Htile[(16 + mb + g) * CC + ncol] = c1[g] + bias;
    }
  }
  __syncthreads();

  // layernorm over C=128 per row; 8 lanes per row
  {
    int row = t >> 3, seg = t & 7;
    float x[16];
    float s1 = 0.f, s2 = 0.f;
#pragma unroll
    for (int j = 0; j < 16; ++j) {
      float v = Htile[row * CC + seg * 16 + j];
      x[j] = v; s1 += v; s2 += v * v;
    }
#pragma unroll
    for (int m = 1; m < 8; m <<= 1) {
      s1 += __shfl_xor(s1, m, 32);
      s2 += __shfl_xor(s2, m, 32);
    }
    float mu = s1 * (1.f / 128.f);
    float var = s2 * (1.f / 128.f) - mu * mu;
    float rstd = rsqrtf(var + 1e-6f);
    Pack16 o;
#pragma unroll
    for (int j = 0; j < 16; ++j) {
      int col = seg * 16 + j;
      float y = (x[j] - mu) * rstd * ln_g[col] + ln_b[col];
      o.h[j] = (_Float16)y;
    }
    float4* dst = (float4*)(h16 + (long)(vox0 + row) * CC + seg * 16);
    dst[0] = o.f4[0];
    dst[1] = o.f4[1];
  }
}

// ---------------- fallback conv (gathers f32 feats, converts in regs) -------
__global__ __launch_bounds__(256) void k_conv_ln_f32(
    const float* __restrict__ feats, const int* __restrict__ nbr,
    const _Float16* __restrict__ wk, const float* __restrict__ conv_b,
    const float* __restrict__ ln_g, const float* __restrict__ ln_b,
    _Float16* __restrict__ h16) {
  __shared__ __align__(16) _Float16 Atile[32 * CC];
  __shared__ __align__(16) float    Htile[32 * CC];

  const int t = threadIdx.x;
  const int lane = t & 31;
  const int w = t >> 5;
  const int vox0 = blockIdx.x * 32;

  v8f c0 = {}; v8f c1 = {};
  const int grow = t >> 3;
  const int gseg = t & 7;

  for (int off = 0; off < 27; ++off) {
    __syncthreads();
    int gi = nbr[off * NV + vox0 + grow];
    Pack16 pk;
    if (gi >= 0) {
      const float4* src = (const float4*)(feats + (long)gi * CC + gseg * 16);
#pragma unroll
      for (int q = 0; q < 4; ++q) {
        float4 f = src[q];
        pk.h[q * 4 + 0] = (_Float16)f.x; pk.h[q * 4 + 1] = (_Float16)f.y;
        pk.h[q * 4 + 2] = (_Float16)f.z; pk.h[q * 4 + 3] = (_Float16)f.w;
      }
    } else {
      pk.f4[0] = make_float4(0.f, 0.f, 0.f, 0.f);
      pk.f4[1] = make_float4(0.f, 0.f, 0.f, 0.f);
    }
    float4* adst = (float4*)&Atile[grow * CC + gseg * 16];
    adst[0] = pk.f4[0];
    adst[1] = pk.f4[1];
    __syncthreads();

    const _Float16* wko = wk + off * (CC * CC);
    const int arow = lane & 15;
    const int kh = (lane >> 4) * 8;
#pragma unroll
    for (int ks = 0; ks < 4; ++ks) {
      v16h b  = load16(wko + (ks * 32 + kh) * CC + w * 16 + (lane & 15), CC);
      v16h a0 = loadA(&Atile[arow * CC + ks * 32 + kh]);
      v16h a1 = loadA(&Atile[(arow + 16) * CC + ks * 32 + kh]);
      c0 = wmma16(a0, b, c0);
      c1 = wmma16(a1, b, c1);
    }
  }

  {
    int ncol = w * 16 + (lane & 15);
    int mb = (lane >> 4) * 8;
    float bias = conv_b[ncol];
#pragma unroll
    for (int g = 0; g < 8; ++g) {
      Htile[(mb + g) * CC + ncol]      = c0[g] + bias;
      Htile[(16 + mb + g) * CC + ncol] = c1[g] + bias;
    }
  }
  __syncthreads();

  {
    int row = t >> 3, seg = t & 7;
    float x[16];
    float s1 = 0.f, s2 = 0.f;
#pragma unroll
    for (int j = 0; j < 16; ++j) {
      float v = Htile[row * CC + seg * 16 + j];
      x[j] = v; s1 += v; s2 += v * v;
    }
#pragma unroll
    for (int m = 1; m < 8; m <<= 1) {
      s1 += __shfl_xor(s1, m, 32);
      s2 += __shfl_xor(s2, m, 32);
    }
    float mu = s1 * (1.f / 128.f);
    float var = s2 * (1.f / 128.f) - mu * mu;
    float rstd = rsqrtf(var + 1e-6f);
    Pack16 o;
#pragma unroll
    for (int j = 0; j < 16; ++j) {
      int col = seg * 16 + j;
      float y = (x[j] - mu) * rstd * ln_g[col] + ln_b[col];
      o.h[j] = (_Float16)y;
    }
    float4* dst = (float4*)(h16 + (long)(vox0 + row) * CC + seg * 16);
    dst[0] = o.f4[0];
    dst[1] = o.f4[1];
  }
}

// ---------------- MLP: silu(h@w1+b1)@w2+b2 + feats ----------------
__global__ __launch_bounds__(256) void k_mlp(
    const _Float16* __restrict__ h16, const _Float16* __restrict__ w1h,
    const float* __restrict__ b1, const _Float16* __restrict__ w2h,
    const float* __restrict__ b2, const float* __restrict__ feats,
    float* __restrict__ out) {
  __shared__ __align__(16) unsigned char smem[8192 + 32768];
  _Float16* Atile = (_Float16*)smem;            // 32x128 f16
  _Float16* Hid   = (_Float16*)(smem + 8192);   // 32x512 f16
  float*    Stage = (float*)(smem + 8192);      // 32x128 f32 (reuses Hid)

  const int t = threadIdx.x;
  const int lane = t & 31;
  const int w = t >> 5;
  const int vox0 = blockIdx.x * 32;
  const int arow = lane & 15;
  const int kh = (lane >> 4) * 8;
  const int mb = (lane >> 4) * 8;

  // stage A tile via async memory->LDS
  {
    const _Float16* src = h16 + (long)vox0 * CC + t * 16;
    _Float16* dst = Atile + t * 16;
    async_copy_b128(src, dst);
    async_copy_b128(src + 8, dst + 8);
  }
  wait_async0();
  __syncthreads();

  // GEMM1: [32,128] x [128,512]; wave owns 4 N-tiles
  v8f c[2][4];
#pragma unroll
  for (int j = 0; j < 4; ++j) { c[0][j] = (v8f){}; c[1][j] = (v8f){}; }
#pragma unroll
  for (int ks = 0; ks < 4; ++ks) {
    v16h a0 = loadA(&Atile[arow * CC + ks * 32 + kh]);
    v16h a1 = loadA(&Atile[(arow + 16) * CC + ks * 32 + kh]);
#pragma unroll
    for (int j = 0; j < 4; ++j) {
      v16h b = load16(w1h + (ks * 32 + kh) * HD + (w * 4 + j) * 16 + (lane & 15), HD);
      c[0][j] = wmma16(a0, b, c[0][j]);
      c[1][j] = wmma16(a1, b, c[1][j]);
    }
  }

  // bias + silu -> Hid (f16, A layout for GEMM2)
#pragma unroll
  for (int j = 0; j < 4; ++j) {
    int col = (w * 4 + j) * 16 + (lane & 15);
    float bb = b1[col];
#pragma unroll
    for (int g = 0; g < 8; ++g) {
      float x0 = c[0][j][g] + bb;
      float x1 = c[1][j][g] + bb;
      Hid[(mb + g) * HD + col]      = (_Float16)(x0 / (1.f + __expf(-x0)));
      Hid[(16 + mb + g) * HD + col] = (_Float16)(x1 / (1.f + __expf(-x1)));
    }
  }
  __syncthreads();

  // GEMM2: [32,512] x [512,128]; wave owns 1 N-tile
  v8f d0 = {}; v8f d1 = {};
#pragma unroll
  for (int ks = 0; ks < 16; ++ks) {
    v16h a0 = loadA(&Hid[arow * HD + ks * 32 + kh]);
    v16h a1 = loadA(&Hid[(arow + 16) * HD + ks * 32 + kh]);
    v16h b  = load16(w2h + (ks * 32 + kh) * CC + w * 16 + (lane & 15), CC);
    d0 = wmma16(a0, b, d0);
    d1 = wmma16(a1, b, d1);
  }
  __syncthreads();  // Hid reads done before Stage overwrites

  {
    int ncol = w * 16 + (lane & 15);
    float bb = b2[ncol];
#pragma unroll
    for (int g = 0; g < 8; ++g) {
      Stage[(mb + g) * CC + ncol]      = d0[g] + bb;
      Stage[(16 + mb + g) * CC + ncol] = d1[g] + bb;
    }
  }
  __syncthreads();

  // residual + coalesced store
  {
    const float4* fsrc = (const float4*)(feats + (long)vox0 * CC);
    const float4* st4  = (const float4*)Stage;
    float4* dsto = (float4*)(out + (long)vox0 * CC);
#pragma unroll
    for (int q = 0; q < 4; ++q) {
      float4 s = st4[t * 4 + q];
      float4 f = fsrc[t * 4 + q];
      dsto[t * 4 + q] = make_float4(s.x + f.x, s.y + f.y, s.z + f.z, s.w + f.w);
    }
  }
}

extern "C" void kernel_launch(void* const* d_in, const int* in_sizes, int n_in,
                              void* d_out, int out_size, void* d_ws, size_t ws_size,
                              hipStream_t stream) {
  const float* feats  = (const float*)d_in[0];
  const int*   nbr    = (const int*)d_in[1];
  const float* conv_w = (const float*)d_in[2];
  const float* conv_b = (const float*)d_in[3];
  const float* ln_g   = (const float*)d_in[4];
  const float* ln_b   = (const float*)d_in[5];
  const float* w1     = (const float*)d_in[6];
  const float* b1     = (const float*)d_in[7];
  const float* w2     = (const float*)d_in[8];
  const float* b2     = (const float*)d_in[9];
  float* out = (float*)d_out;

  unsigned char* ws = (unsigned char*)d_ws;
  const size_t SZ_H16 = (size_t)NV * CC * 2;          // 64 MB
  _Float16* h16     = (_Float16*)ws;
  _Float16* wk      = (_Float16*)(ws + SZ_H16);
  _Float16* w1h     = (_Float16*)(ws + SZ_H16 + 884736);
  _Float16* w2h     = (_Float16*)(ws + SZ_H16 + 884736 + 131072);
  _Float16* feats16 = (_Float16*)(ws + SZ_H16 + 884736 + 262144);

  const size_t need_async = SZ_H16 + 884736 + 262144 + SZ_H16;  // ~129 MB
  const bool use_async = (ws_size >= need_async);

  const int prep_elems = 27 * CC * CC + 2 * CC * HD;
  k_prep<<<(prep_elems + 255) / 256, 256, 0, stream>>>(conv_w, w1, w2, wk, w1h, w2h);

  if (use_async) {
    k_feats16<<<(NV * CC) / (256 * 16), 256, 0, stream>>>(feats, feats16);
    k_conv_ln_async<<<NV / 32, 256, 0, stream>>>(feats16, nbr, wk, conv_b,
                                                 ln_g, ln_b, h16);
  } else {
    k_conv_ln_f32<<<NV / 32, 256, 0, stream>>>(feats, nbr, wk, conv_b,
                                               ln_g, ln_b, h16);
  }
  k_mlp<<<NV / 32, 256, 0, stream>>>(h16, w1h, b1, w2h, b2, feats, out);
}
